// MultiHeadAttention_7301444403901
// MI455X (gfx1250) — matrix operand order repaired and checked by reading, not compiled, not run on hardware
//
#include <hip/hip_runtime.h>

#define DMODEL 1024
#define SEQ    2048
#define NHEAD  16
#define HDIM   64

typedef __attribute__((ext_vector_type(16))) _Float16 v16h;
typedef __attribute__((ext_vector_type(8)))  _Float16 v8h;
typedef __attribute__((ext_vector_type(8)))  float    v8f;
typedef __attribute__((ext_vector_type(4)))  float    v4f;

union Frag16 { v16h v; v8h h[2]; };

__device__ __forceinline__ v8f wmma_f16(const Frag16& a, const Frag16& b, v8f c) {
  // D = A(16x32 f16) * B(32x16 f16) + C(16x16 f32)
  return __builtin_amdgcn_wmma_f32_16x16x32_f16(false, a.v, false, b.v,
                                                (short)0, c, false, false);
}

__device__ __forceinline__ v8h cvt8(v4f a, v4f b) {
  v8h h;
  h[0] = (_Float16)a[0]; h[1] = (_Float16)a[1];
  h[2] = (_Float16)a[2]; h[3] = (_Float16)a[3];
  h[4] = (_Float16)b[0]; h[5] = (_Float16)b[1];
  h[6] = (_Float16)b[2]; h[7] = (_Float16)b[3];
  return h;
}

// CDNA5 async copy: global -> LDS, 16B per lane, tracked by ASYNCcnt.
__device__ __forceinline__ void async_ld_b128(unsigned lds_off, const void* gaddr) {
  asm volatile("global_load_async_to_lds_b128 %0, %1, off"
               :: "v"(lds_off), "v"((unsigned long long)(uintptr_t)gaddr)
               : "memory");
}
__device__ __forceinline__ void wait_asynccnt0() {
  asm volatile("s_wait_asynccnt 0" ::: "memory");
}

// C[m,n] = sum_k X[m,k] * W[n,k] + bias[n]      (i.e. X @ W^T + b)
// 128 threads = 4 waves; block tile 128x128; wave tile 64x64; K-step 32.
// Double-buffered LDS, software-pipelined staging (loads issue before WMMAs).
template<typename TX, bool OUTF32>
__global__ __launch_bounds__(128) void mha_gemm_xwT(
    const TX* __restrict__ X, const float* __restrict__ W,
    const float* __restrict__ bias, _Float16* __restrict__ outH,
    float* __restrict__ outF, int M, int N, int K)
{
  __shared__ __align__(16) _Float16 sA[2][128 * 32];
  __shared__ __align__(16) _Float16 sB[2][128 * 32];

  const int tid  = threadIdx.x;
  const int wave = tid >> 5, lane = tid & 31;
  const int lm = lane & 15, lg = lane >> 4;
  const int bm = blockIdx.y * 128, bn = blockIdx.x * 128;
  const int wm = (wave >> 1) * 64, wn = (wave & 1) * 64;

  int srow[4], scol[4];
  #pragma unroll
  for (int i = 0; i < 4; ++i) {
    int seg = tid + i * 128;        // 0..511
    srow[i] = seg >> 2;             // 0..127
    scol[i] = (seg & 3) * 8;        // 0,8,16,24
  }

  v8f acc[4][4] = {};

  // Staging registers (loads issued early, consumed at storeTile).
  v4f aLo[4], aHi[4], bLo[4], bHi[4];   // f32 path
  v8h aH[4];                            // f16 X path

  auto loadTile = [&](int k0) {
    #pragma unroll
    for (int i = 0; i < 4; ++i) {
      const size_t xo = (size_t)(bm + srow[i]) * K + k0 + scol[i];
      if constexpr (sizeof(TX) == 2) {
        aH[i] = *(const v8h*)(X + xo);
      } else {
        aLo[i] = *(const v4f*)(X + xo);
        aHi[i] = *(const v4f*)(X + xo + 4);
      }
      const size_t wo = (size_t)(bn + srow[i]) * K + k0 + scol[i];
      bLo[i] = *(const v4f*)(W + wo);
      bHi[i] = *(const v4f*)(W + wo + 4);
    }
  };
  auto storeTile = [&](int buf) {
    #pragma unroll
    for (int i = 0; i < 4; ++i) {
      _Float16* da = &sA[buf][srow[i] * 32 + scol[i]];
      if constexpr (sizeof(TX) == 2) *(v8h*)da = aH[i];
      else                           *(v8h*)da = cvt8(aLo[i], aHi[i]);
      *(v8h*)&sB[buf][srow[i] * 32 + scol[i]] = cvt8(bLo[i], bHi[i]);
    }
  };

  loadTile(0);
  storeTile(0);
  __syncthreads();

  for (int k0 = 0; k0 < K; k0 += 32) {
    const int  cur     = (k0 >> 5) & 1;
    const bool hasNext = (k0 + 32) < K;
    if (hasNext) loadTile(k0 + 32);           // global loads in flight over WMMAs

    Frag16 af[4], bf[4];
    #pragma unroll
    for (int mt = 0; mt < 4; ++mt) {          // A-frag: lane = row lm, K per lg
      const _Float16* p = &sA[cur][(wm + mt * 16 + lm) * 32 + lg * 8];
      af[mt].h[0] = *(const v8h*)p;           // K 0-7   / 8-15
      af[mt].h[1] = *(const v8h*)(p + 16);    // K 16-23 / 24-31
    }
    #pragma unroll
    for (int nt = 0; nt < 4; ++nt) {          // B-frag: lane n = W row n, K per lg
      const _Float16* p = &sB[cur][(wn + nt * 16 + lm) * 32 + lg * 8];
      bf[nt].h[0] = *(const v8h*)p;
      bf[nt].h[1] = *(const v8h*)(p + 16);
    }
    #pragma unroll
    for (int mt = 0; mt < 4; ++mt)
      #pragma unroll
      for (int nt = 0; nt < 4; ++nt)
        acc[mt][nt] = wmma_f16(af[mt], bf[nt], acc[mt][nt]);

    if (hasNext) storeTile(cur ^ 1);          // waits for loads here, not earlier
    __syncthreads();
  }

  // Epilogue: C layout lane L -> (row = 8*lg + v, col = lm) within a 16x16 tile.
  #pragma unroll
  for (int nt = 0; nt < 4; ++nt) {
    int col = bn + wn + nt * 16 + lm;
    float bv = bias[col];
    #pragma unroll
    for (int mt = 0; mt < 4; ++mt) {
      int row0 = bm + wm + mt * 16 + lg * 8;
      #pragma unroll
      for (int v = 0; v < 8; ++v) {
        float val = acc[mt][nt][v] + bv;
        if constexpr (OUTF32) outF[(size_t)(row0 + v) * N + col] = val;
        else                  outH[(size_t)(row0 + v) * N + col] = (_Float16)val;
      }
    }
  }
}

// Flash attention: block = (128 q-rows, head, batch); 4 waves x 32 q-rows each
// (2 M-tiles per wave so every K/V^T B-fragment feeds two WMMAs -> 2x the
// FLOP per LDS byte). K-tiles copied with GLOBAL_LOAD_ASYNC_TO_LDS_B128
// (ASYNCcnt); V transposed through VGPRs. sK/sVT double-buffered.
// scale = 1/sqrt(D_MODEL) = 1/32 (reference scales by d_model, not head_dim).
__global__ __launch_bounds__(128) void mha_flash(
    const _Float16* __restrict__ Qh, const _Float16* __restrict__ Kh,
    const _Float16* __restrict__ Vh, _Float16* __restrict__ Oh)
{
  __shared__ __align__(16) _Float16 sK [2][64 * 64];   // [key][dh]
  __shared__ __align__(16) _Float16 sVT[2][64 * 64];   // [dh][key]
  __shared__ __align__(16) _Float16 sP [4][32 * 64];   // per-wave P tile

  const int tid  = threadIdx.x;
  const int wave = tid >> 5, lane = tid & 31;
  const int lm = lane & 15, lg = lane >> 4;
  const int h  = blockIdx.y;
  const size_t rowbase = (size_t)blockIdx.z * SEQ;
  const int q0 = blockIdx.x * 128 + wave * 32;
  const size_t hoff = (size_t)h * HDIM;
  const int NB = SEQ / 64;

  int srow[4], scol[4];
  #pragma unroll
  for (int i = 0; i < 4; ++i) {
    int seg = tid + i * 128;        // 0..511
    srow[i] = seg >> 3;             // 0..63 (key)
    scol[i] = (seg & 7) * 8;        // 0..56 (dh)
  }

  v8h vR[4];
  auto loadKV = [&](int kb, int buf) {
    #pragma unroll
    for (int i = 0; i < 4; ++i) {
      const size_t go = (rowbase + kb * 64 + srow[i]) * DMODEL + hoff + scol[i];
      async_ld_b128((unsigned)(uintptr_t)&sK[buf][srow[i] * 64 + scol[i]], Kh + go);
      vR[i] = *(const v8h*)(Vh + go);
    }
  };
  auto storeV = [&](int buf) {
    #pragma unroll
    for (int i = 0; i < 4; ++i)
      #pragma unroll
      for (int e = 0; e < 8; ++e)
        sVT[buf][(scol[i] + e) * 64 + srow[i]] = vR[i][e];
  };

  // Q A-fragments: 2 M-tiles x (16 rows x 64 dh) kept in registers.
  Frag16 qf[2][2];
  #pragma unroll
  for (int mt = 0; mt < 2; ++mt)
    #pragma unroll
    for (int ks = 0; ks < 2; ++ks) {
      const _Float16* p =
          Qh + (rowbase + q0 + mt * 16 + lm) * DMODEL + hoff + ks * 32 + lg * 8;
      qf[mt][ks].h[0] = *(const v8h*)p;
      qf[mt][ks].h[1] = *(const v8h*)(p + 16);
    }

  float m_i[2][8], l_i[2][8];
  #pragma unroll
  for (int mt = 0; mt < 2; ++mt)
    #pragma unroll
    for (int v = 0; v < 8; ++v) { m_i[mt][v] = -3.0e38f; l_i[mt][v] = 0.f; }
  v8f oacc[2][4] = {};

  loadKV(0, 0);
  storeV(0);
  wait_asynccnt0();
  __syncthreads();

  for (int kb = 0; kb < NB; ++kb) {
    const int  cur     = kb & 1;
    const bool hasNext = (kb + 1) < NB;
    if (hasNext) loadKV(kb + 1, cur ^ 1);    // async K + V-loads over compute

    // Logits: S = Q @ K^T; each K-fragment shared by both M-tiles.
    v8f st[2][4];
    #pragma unroll
    for (int nt = 0; nt < 4; ++nt) {
      Frag16 kf[2];
      #pragma unroll
      for (int ks = 0; ks < 2; ++ks) {
        const _Float16* p = &sK[cur][(nt * 16 + lm) * 64 + ks * 32 + lg * 8];
        kf[ks].h[0] = *(const v8h*)p;
        kf[ks].h[1] = *(const v8h*)(p + 16);
      }
      #pragma unroll
      for (int mt = 0; mt < 2; ++mt) {
        v8f c = {};
        c = wmma_f16(qf[mt][0], kf[0], c);
        c = wmma_f16(qf[mt][1], kf[1], c);
        st[mt][nt] = c;
      }
    }

    // Online softmax (row m = mt*16 + lg*8 + v; reduce across the 16 lm-lanes).
    #pragma unroll
    for (int mt = 0; mt < 2; ++mt)
      #pragma unroll
      for (int v = 0; v < 8; ++v) {
        float r = -3.0e38f;
        #pragma unroll
        for (int nt = 0; nt < 4; ++nt) {
          st[mt][nt][v] *= 0.03125f;           // 1/sqrt(1024)
          r = fmaxf(r, st[mt][nt][v]);
        }
        #pragma unroll
        for (int off = 8; off > 0; off >>= 1) r = fmaxf(r, __shfl_xor(r, off, 32));
        float nm    = fmaxf(m_i[mt][v], r);
        float alpha = __expf(m_i[mt][v] - nm);
        m_i[mt][v] = nm;
        float rs = 0.f;
        #pragma unroll
        for (int nt = 0; nt < 4; ++nt) {
          float p = __expf(st[mt][nt][v] - nm);
          st[mt][nt][v] = p;
          rs += p;
        }
        #pragma unroll
        for (int off = 8; off > 0; off >>= 1) rs += __shfl_xor(rs, off, 32);
        l_i[mt][v] = l_i[mt][v] * alpha + rs;
        #pragma unroll
        for (int dt = 0; dt < 4; ++dt) oacc[mt][dt][v] *= alpha;
      }

    // P: C-layout f32 -> A-layout f16 via per-wave LDS round trip.
    // Same-wave DS ops are in-order; no block barrier needed (private region).
    _Float16* myP = &sP[wave][0];
    #pragma unroll
    for (int mt = 0; mt < 2; ++mt)
      #pragma unroll
      for (int nt = 0; nt < 4; ++nt)
        #pragma unroll
        for (int v = 0; v < 8; ++v)
          myP[(mt * 16 + lg * 8 + v) * 64 + nt * 16 + lm] = (_Float16)st[mt][nt][v];

    Frag16 pf[2][2];
    #pragma unroll
    for (int mt = 0; mt < 2; ++mt)
      #pragma unroll
      for (int ks = 0; ks < 2; ++ks) {
        const _Float16* p = &myP[(mt * 16 + lm) * 64 + ks * 32 + lg * 8];
        pf[mt][ks].h[0] = *(const v8h*)p;
        pf[mt][ks].h[1] = *(const v8h*)(p + 16);
      }
    // O += P @ V; each V^T fragment shared by both M-tiles.
    #pragma unroll
    for (int dt = 0; dt < 4; ++dt) {
      #pragma unroll
      for (int ks = 0; ks < 2; ++ks) {
        Frag16 vf;
        const _Float16* p = &sVT[cur][(dt * 16 + lm) * 64 + ks * 32 + lg * 8];
        vf.h[0] = *(const v8h*)p;
        vf.h[1] = *(const v8h*)(p + 16);
        #pragma unroll
        for (int mt = 0; mt < 2; ++mt)
          oacc[mt][dt] = wmma_f16(pf[mt][ks], vf, oacc[mt][dt]);
      }
    }

    if (hasNext) storeV(cur ^ 1);            // V transpose scatter into next buf
    wait_asynccnt0();                        // async K-tile landed in LDS
    __syncthreads();
  }

  // Normalize and store attention output (f16, [B*S, D] with head offset).
  #pragma unroll
  for (int mt = 0; mt < 2; ++mt)
    #pragma unroll
    for (int v = 0; v < 8; ++v) {
      float inv = 1.0f / l_i[mt][v];
      size_t row = (rowbase + q0 + mt * 16 + lg * 8 + v) * DMODEL + hoff;
      #pragma unroll
      for (int dt = 0; dt < 4; ++dt)
        Oh[row + dt * 16 + lm] = (_Float16)(oacc[mt][dt][v] * inv);
    }
}

extern "C" void kernel_launch(void* const* d_in, const int* in_sizes, int n_in,
                              void* d_out, int out_size, void* d_ws, size_t ws_size,
                              hipStream_t stream) {
  const float* q  = (const float*)d_in[0];
  const float* k  = (const float*)d_in[1];
  const float* v  = (const float*)d_in[2];
  const float* Wq = (const float*)d_in[3];
  const float* bq = (const float*)d_in[4];
  const float* Wk = (const float*)d_in[5];
  const float* bk = (const float*)d_in[6];
  const float* Wv = (const float*)d_in[7];
  const float* bv = (const float*)d_in[8];
  const float* Wo = (const float*)d_in[9];
  const float* bo = (const float*)d_in[10];

  const int B = in_sizes[0] / (SEQ * DMODEL);     // = 2
  const int M = B * SEQ;                          // 4096 rows
  const size_t BSD = (size_t)M * DMODEL;

  // Workspace: 4 f16 buffers of B*S*D halves (8 MB each, 32 MB total).
  _Float16* Qh = (_Float16*)d_ws;
  _Float16* Kh = Qh + BSD;
  _Float16* Vh = Kh + BSD;
  _Float16* Ah = Vh + BSD;

  dim3 blk(128);
  dim3 g(DMODEL / 128, M / 128);                  // (8, 32)
  mha_gemm_xwT<float, false><<<g, blk, 0, stream>>>(q, Wq, bq, Qh, nullptr, M, DMODEL, DMODEL);
  mha_gemm_xwT<float, false><<<g, blk, 0, stream>>>(k, Wk, bk, Kh, nullptr, M, DMODEL, DMODEL);
  mha_gemm_xwT<float, false><<<g, blk, 0, stream>>>(v, Wv, bv, Vh, nullptr, M, DMODEL, DMODEL);

  dim3 ga(SEQ / 128, NHEAD, B);                   // (16, 16, 2)
  mha_flash<<<ga, blk, 0, stream>>>(Qh, Kh, Vh, Ah);

  mha_gemm_xwT<_Float16, true><<<g, blk, 0, stream>>>(Ah, Wo, bo, nullptr, (float*)d_out, M, DMODEL, DMODEL);
}
